// Attn_17738214933129
// MI455X (gfx1250) — matrix operand order repaired and checked by reading, not compiled, not run on hardware
//
#include <hip/hip_runtime.h>

// Problem dims
#define Bq    16
#define Sq    512
#define NLOCq 2048
#define FEATq 512
#define Hq    8
#define DHq   64
#define Mq    (Bq * Sq)   // 8192 rows of (b,s)

typedef __attribute__((ext_vector_type(16))) __bf16 bf16x16;
typedef __attribute__((ext_vector_type(8)))  __bf16 bf16x8;
typedef __attribute__((ext_vector_type(8)))  float  f32x8;

static __device__ __forceinline__ f32x8 zero8() {
  f32x8 z;
#pragma unroll
  for (int i = 0; i < 8; ++i) z[i] = 0.0f;
  return z;
}

static __device__ __forceinline__ f32x8 wmma_bf16(bf16x16 a, bf16x16 b, f32x8 c) {
  // D = A(16x32 bf16) * B(32x16 bf16) + C(16x16 f32)
  return __builtin_amdgcn_wmma_f32_16x16x32_bf16(
      /*neg_a=*/false, a, /*neg_b=*/false, b,
      /*c_mod=*/(short)0, c, /*reuse_a=*/false, /*reuse_b=*/false);
}

// Build an A-fragment (16x32 bf16, contiguous k order, same as B):
// lane L -> row m = L%16 ; elements 0..15 = K[c0..c0+15]
// where c0 = (L<16 ? 0 : 16).
static __device__ __forceinline__ bf16x16 load_afrag(const __bf16* p /* row base + c0 */) {
  bf16x8 lo = *reinterpret_cast<const bf16x8*>(p);
  bf16x8 hb = *reinterpret_cast<const bf16x8*>(p + 8);
  return __builtin_shufflevector(lo, hb, 0, 1, 2, 3, 4, 5, 6, 7,
                                 8, 9, 10, 11, 12, 13, 14, 15);
}

static __device__ __forceinline__ bf16x8 cvt8(f32x8 v) {
  bf16x8 b;
#pragma unroll
  for (int e = 0; e < 8; ++e) b[e] = (__bf16)v[e];
  return b;
}

// ---------------------------------------------------------------------------
// f32 -> bf16 elementwise (weights)
// ---------------------------------------------------------------------------
__global__ void cvt_f32_bf16(const float* __restrict__ in, __bf16* __restrict__ out, int n) {
  int i = blockIdx.x * 256 + threadIdx.x;
  if (i < n) out[i] = (__bf16)in[i];
}

// ---------------------------------------------------------------------------
// sm = softmax(where(mask==0, -1e9, str_mat), axis=-1)   rows = B*S, cols = S
// one wave per row (wave32: 16 elements/lane)
// ---------------------------------------------------------------------------
__global__ __launch_bounds__(256) void sm_softmax_k(const float* __restrict__ str_mat,
                                                    const int* __restrict__ mask,
                                                    float* __restrict__ sm_out) {
  const int row  = blockIdx.x * 8 + (threadIdx.x >> 5);
  const int lane = threadIdx.x & 31;
  const size_t base = (size_t)row * Sq;
  float x[16];
  float mx = -3.0e38f;
#pragma unroll
  for (int e = 0; e < 16; ++e) {
    const int c = e * 32 + lane;
    const float v = (mask[base + c] == 0) ? -1.0e9f : str_mat[base + c];
    x[e] = v;
    mx = fmaxf(mx, v);
  }
#pragma unroll
  for (int off = 16; off; off >>= 1) mx = fmaxf(mx, __shfl_xor(mx, off, 32));
  float sum = 0.0f;
#pragma unroll
  for (int e = 0; e < 16; ++e) { x[e] = __expf(x[e] - mx); sum += x[e]; }
#pragma unroll
  for (int off = 16; off; off >>= 1) sum += __shfl_xor(sum, off, 32);
  const float inv = 1.0f / sum;
#pragma unroll
  for (int e = 0; e < 16; ++e) sm_out[base + e * 32 + lane] = x[e] * inv;
}

// ---------------------------------------------------------------------------
// Generic GEMM: Y[m][n] = sum_k X[m][k] * W[n][k] + bias[n]
//   X: f32 (M x K) row-major, converted to bf16 while double-buffer staging
//      into LDS (next tile's global load overlaps current tile's 16 WMMAs)
//   W: bf16 (N x K) row-major  -> rows are exactly WMMA B-fragments
// Block: 256 thr (8 waves). Block tile M=64 x N=512; wave tile 64x64 (16 acc).
//   mode 0: out bf16 Q/K layout [b,h,s,d]
//   mode 1: out bf16 V-transposed layout [b,h,d,s]
//   mode 2: out f32 [m][NLOC] (final projection; grid.y tiles N)
// ---------------------------------------------------------------------------
__global__ __launch_bounds__(256) void gemm_xwt(const float* __restrict__ X,
                                                const __bf16* __restrict__ W,
                                                const float* __restrict__ bias,
                                                __bf16* __restrict__ outb,
                                                float* __restrict__ outf,
                                                int K, int mode) {
  // 80B-padded rows: A-fragment ds_load_b128 starts are lane*20 dwords (mod 64)
  // -> 16 distinct multiples of 4 -> conflict-free. Buffer stride 5120B == 0 mod
  // bank period, so both buffers have the same conflict-free pattern.
  __shared__ __align__(16) __bf16 sA[2][64][40];

  const int w    = threadIdx.x >> 5;
  const int lane = threadIdx.x & 31;
  const int lc   = lane & 15;
  const int hh   = lane >> 4;         // 0 = low half, 1 = high half
  const int m0   = blockIdx.x * 64;
  const int n0   = blockIdx.y * 512 + w * 64;

  f32x8 acc[4][4];
#pragma unroll
  for (int a = 0; a < 4; ++a)
#pragma unroll
    for (int c = 0; c < 4; ++c) acc[a][c] = zero8();

  // staging assignment: thread t loads 8 f32 from row (t>>2), k-offset (t&3)*8
  const int rs  = threadIdx.x >> 2;
  const int cs8 = (threadIdx.x & 3) * 8;
  const size_t xrow = (size_t)(m0 + rs) * (size_t)K;

  // prologue: stage tile 0
  *reinterpret_cast<bf16x8*>(&sA[0][rs][cs8]) =
      cvt8(*reinterpret_cast<const f32x8*>(X + xrow + cs8));
  __syncthreads();

  int cur = 0;
#pragma unroll 1
  for (int kk = 0; kk < K; kk += 32) {
    const bool nxt = (kk + 32) < K;
    f32x8 vn = zero8();
    if (nxt)  // issue next tile's global load before the WMMA chain
      vn = *reinterpret_cast<const f32x8*>(X + xrow + kk + 32 + cs8);

    bf16x16 af[4], bfr[4];
#pragma unroll
    for (int t = 0; t < 4; ++t)
      af[t] = load_afrag(&sA[cur][t * 16 + lc][hh * 16]);
#pragma unroll
    for (int t = 0; t < 4; ++t) {
      const __bf16* bp = W + (size_t)(n0 + t * 16 + lc) * (size_t)K + kk + hh * 16;
      bfr[t] = *reinterpret_cast<const bf16x16*>(bp);
    }
#pragma unroll
    for (int tm = 0; tm < 4; ++tm)
#pragma unroll
      for (int tn = 0; tn < 4; ++tn)
        acc[tm][tn] = wmma_bf16(af[tm], bfr[tn], acc[tm][tn]);

    if (nxt)
      *reinterpret_cast<bf16x8*>(&sA[cur ^ 1][rs][cs8]) = cvt8(vn);
    __syncthreads();
    cur ^= 1;
  }

  // C layout: lane holds col n = 16-tile + lc, VGPR i -> row m = tile + i + 8*hh
  const int lm = hh * 8;
#pragma unroll
  for (int tm = 0; tm < 4; ++tm)
#pragma unroll
    for (int tn = 0; tn < 4; ++tn) {
      const int n  = n0 + tn * 16 + lc;
      const float bv = bias[n];
#pragma unroll
      for (int i = 0; i < 8; ++i) {
        const int   m   = m0 + tm * 16 + lm + i;
        const float val = acc[tm][tn][i] + bv;
        const int bb = m >> 9, s = m & 511;
        if (mode == 0) {
          const int h = n >> 6, d = n & 63;
          outb[((size_t)(bb * Hq + h) * Sq + s) * DHq + d] = (__bf16)val;
        } else if (mode == 1) {
          const int h = n >> 6, d = n & 63;
          outb[((size_t)(bb * Hq + h) * DHq + d) * Sq + s] = (__bf16)val;
        } else {
          outf[(size_t)m * NLOCq + n] = val;
        }
      }
    }
}

// ---------------------------------------------------------------------------
// P = softmax(Q K^T / DH + sm)  per (b,h), 32 query rows per block, bf16 out.
// Q,K: bf16 [b,h,s,d]. 8 waves: wave w computes cols [w*64, w*64+64).
// ---------------------------------------------------------------------------
__global__ __launch_bounds__(256) void attn_scores_k(const __bf16* __restrict__ Q,
                                                     const __bf16* __restrict__ Kb,
                                                     const float* __restrict__ smb,
                                                     __bf16* __restrict__ P) {
  __shared__ __align__(16) float sc[32][512];   // 64 KB score tile

  const int b = blockIdx.z, h = blockIdx.y;
  const int m0 = blockIdx.x * 32;
  const int w = threadIdx.x >> 5, lane = threadIdx.x & 31;
  const int lc = lane & 15, hh = lane >> 4;
  const size_t head = (size_t)(b * Hq + h) * Sq;
  const __bf16* Qh = Q + head * DHq;
  const __bf16* Kh = Kb + head * DHq;
  const int n0 = w * 64;

  f32x8 acc[2][4];
#pragma unroll
  for (int a = 0; a < 2; ++a)
#pragma unroll
    for (int c = 0; c < 4; ++c) acc[a][c] = zero8();

#pragma unroll
  for (int kk = 0; kk < DHq; kk += 32) {
    bf16x16 af[2], bfr[4];
#pragma unroll
    for (int t = 0; t < 2; ++t)
      af[t] = load_afrag(Qh + (size_t)(m0 + t * 16 + lc) * DHq + kk + hh * 16);
#pragma unroll
    for (int t = 0; t < 4; ++t)
      bfr[t] = *reinterpret_cast<const bf16x16*>(
          Kh + (size_t)(n0 + t * 16 + lc) * DHq + kk + hh * 16);
#pragma unroll
    for (int tm = 0; tm < 2; ++tm)
#pragma unroll
      for (int tn = 0; tn < 4; ++tn)
        acc[tm][tn] = wmma_bf16(af[tm], bfr[tn], acc[tm][tn]);
  }

  // scores -> LDS with 1/DH scale + structural-softmax bias
  const float scale = 1.0f / (float)DHq;
#pragma unroll
  for (int tm = 0; tm < 2; ++tm)
#pragma unroll
    for (int tn = 0; tn < 4; ++tn) {
      const int n = n0 + tn * 16 + lc;
#pragma unroll
      for (int i = 0; i < 8; ++i) {
        const int rloc = tm * 16 + hh * 8 + i;
        const int mg   = m0 + rloc;
        sc[rloc][n] = acc[tm][tn][i] * scale +
                      smb[((size_t)b * Sq + mg) * Sq + n];
      }
    }
  __syncthreads();

  // row softmax: wave w owns rows w*4 .. w*4+3
#pragma unroll
  for (int j = 0; j < 4; ++j) {
    const int r = w * 4 + j;
    float x[16];
    float mx = -3.0e38f;
#pragma unroll
    for (int e = 0; e < 16; ++e) {
      x[e] = sc[r][e * 32 + lane];
      mx = fmaxf(mx, x[e]);
    }
#pragma unroll
    for (int off = 16; off; off >>= 1) mx = fmaxf(mx, __shfl_xor(mx, off, 32));
    float sum = 0.0f;
#pragma unroll
    for (int e = 0; e < 16; ++e) { x[e] = __expf(x[e] - mx); sum += x[e]; }
#pragma unroll
    for (int off = 16; off; off >>= 1) sum += __shfl_xor(sum, off, 32);
    const float inv = 1.0f / sum;
    __bf16* pr = P + (head + m0 + r) * Sq;
#pragma unroll
    for (int e = 0; e < 16; ++e) pr[e * 32 + lane] = (__bf16)(x[e] * inv);
  }
}

// ---------------------------------------------------------------------------
// x_att[b,s,h*64+d] = sum_k P[b,h,s,k] * V[b,h,k,d]   (V stored transposed)
// Block: 128 query rows, 8 waves -> wave tile 16x64 (4 acc), K=512.
// P is streamed once from HBM: prefetch the next chunk of the A row.
// ---------------------------------------------------------------------------
__global__ __launch_bounds__(256) void attn_pv_k(const __bf16* __restrict__ P,
                                                 const __bf16* __restrict__ Vt,
                                                 float* __restrict__ xatt) {
  const int b = blockIdx.z, h = blockIdx.y;
  const int m0 = blockIdx.x * 128;
  const int w = threadIdx.x >> 5, lane = threadIdx.x & 31;
  const int lc = lane & 15, hh = lane >> 4;
  const size_t head = (size_t)(b * Hq + h);
  const __bf16* Ph = P + head * (size_t)Sq * Sq;
  const __bf16* Vh = Vt + head * (size_t)DHq * Sq;
  const int mw = m0 + w * 16;

  f32x8 acc[4];
#pragma unroll
  for (int c = 0; c < 4; ++c) acc[c] = zero8();

  const __bf16* arow = Ph + (size_t)(mw + lc) * Sq;
#pragma unroll 1
  for (int kk = 0; kk < Sq; kk += 32) {
    if (kk + 64 < Sq) __builtin_prefetch(arow + kk + 64 + hh * 8, 0, 1);
    bf16x16 af = load_afrag(arow + kk + hh * 16);
#pragma unroll
    for (int tn = 0; tn < 4; ++tn) {
      bf16x16 bfr = *reinterpret_cast<const bf16x16*>(
          Vh + (size_t)(tn * 16 + lc) * Sq + kk + hh * 16);
      acc[tn] = wmma_bf16(af, bfr, acc[tn]);
    }
  }

#pragma unroll
  for (int tn = 0; tn < 4; ++tn) {
    const int d = tn * 16 + lc;
#pragma unroll
    for (int i = 0; i < 8; ++i) {
      const int s = mw + hh * 8 + i;
      xatt[((size_t)b * Sq + s) * FEATq + h * DHq + d] = acc[tn][i];
    }
  }
}

// ---------------------------------------------------------------------------
extern "C" void kernel_launch(void* const* d_in, const int* in_sizes, int n_in,
                              void* d_out, int out_size, void* d_ws, size_t ws_size,
                              hipStream_t stream) {
  (void)in_sizes; (void)n_in; (void)out_size; (void)ws_size;
  const float* q   = (const float*)d_in[0];
  const float* k   = (const float*)d_in[1];
  const float* v   = (const float*)d_in[2];
  const float* smi = (const float*)d_in[3];
  const int*   msk = (const int*)d_in[4];
  const float* Wq  = (const float*)d_in[5];
  const float* bq  = (const float*)d_in[6];
  const float* Wk  = (const float*)d_in[7];
  const float* bk  = (const float*)d_in[8];
  const float* Wv  = (const float*)d_in[9];
  const float* bv  = (const float*)d_in[10];
  const float* Wo  = (const float*)d_in[11];
  const float* bo  = (const float*)d_in[12];
  float* out = (float*)d_out;

  char* ws = (char*)d_ws;
  size_t off = 0;
  auto take = [&](size_t bytes) -> char* {
    char* p = ws + off;
    off += bytes;
    off = (off + 255) & ~(size_t)255;
    return p;
  };

  const int WN = FEATq * NLOCq;  // 512*2048 == 2048*512
  __bf16* Wqb = (__bf16*)take((size_t)WN * 2);
  __bf16* Wkb = (__bf16*)take((size_t)WN * 2);
  __bf16* Wvb = (__bf16*)take((size_t)WN * 2);
  __bf16* Wob = (__bf16*)take((size_t)WN * 2);
  float*  smx = (float*)take((size_t)Bq * Sq * Sq * 4);
  __bf16* Qb  = (__bf16*)take((size_t)Mq * FEATq * 2);
  __bf16* Kbf = (__bf16*)take((size_t)Mq * FEATq * 2);
  __bf16* Vtb = (__bf16*)take((size_t)Mq * FEATq * 2);
  __bf16* P   = (__bf16*)take((size_t)Bq * Hq * Sq * Sq * 2);
  float*  xat = (float*)take((size_t)Mq * FEATq * 4);

  const int cgrid = (WN + 255) / 256;
  cvt_f32_bf16<<<cgrid, 256, 0, stream>>>(Wq, Wqb, WN);
  cvt_f32_bf16<<<cgrid, 256, 0, stream>>>(Wk, Wkb, WN);
  cvt_f32_bf16<<<cgrid, 256, 0, stream>>>(Wv, Wvb, WN);
  cvt_f32_bf16<<<cgrid, 256, 0, stream>>>(Wo, Wob, WN);

  sm_softmax_k<<<Mq / 8, 256, 0, stream>>>(smi, msk, smx);

  gemm_xwt<<<dim3(Mq / 64, 1), 256, 0, stream>>>(q, Wqb, bq, Qb,  nullptr, NLOCq, 0);
  gemm_xwt<<<dim3(Mq / 64, 1), 256, 0, stream>>>(k, Wkb, bk, Kbf, nullptr, NLOCq, 0);
  gemm_xwt<<<dim3(Mq / 64, 1), 256, 0, stream>>>(v, Wvb, bv, Vtb, nullptr, NLOCq, 1);

  attn_scores_k<<<dim3(Sq / 32, Hq, Bq), 256, 0, stream>>>(Qb, Kbf, smx, P);
  attn_pv_k<<<dim3(Sq / 128, Hq, Bq), 256, 0, stream>>>(P, Vtb, xat);

  gemm_xwt<<<dim3(Mq / 64, NLOCq / 512), 256, 0, stream>>>(xat, Wob, bo, nullptr, out,
                                                           FEATq, 2);
}
